// GCN_56676388438268
// MI455X (gfx1250) — compile-verified, matrix operand order checked by reading, not run on hardware
//
#include <hip/hip_runtime.h>
#include <hip/hip_bf16.h>
#include <math.h>

// ---------------------------------------------------------------------------
// Types for CDNA5 WMMA (gfx1250, wave32)
// ---------------------------------------------------------------------------
typedef __attribute__((ext_vector_type(16))) __bf16 v16bf;
typedef __attribute__((ext_vector_type(8)))  float  v8f;

#define GCN_N 40000
#define GCN_E 640000
#define GCN_G 256
#define GCN_H 128
#define LDS_K 132   // padded k-stride (halves) to dodge LDS bank conflicts

// ---------------------------------------------------------------------------
// Zero-fill kernel (workspace is poisoned by the harness; re-init every call)
// ---------------------------------------------------------------------------
__global__ void gcn_zero_f32(float* __restrict__ p, int n) {
    int i = blockIdx.x * blockDim.x + threadIdx.x;
    if (i < n) p[i] = 0.0f;
}

// ---------------------------------------------------------------------------
// Edge scatter: agg[dst] += h[src]. One wave32 per edge, float4 per lane.
// The whole feature matrix (20 MB) fits in the 192 MB L2, so these gathers
// and f32 atomic adds run at L2 rates after first touch.
// ---------------------------------------------------------------------------
__global__ void __launch_bounds__(256) gcn_edge_scatter(
    const float* __restrict__ h, const int* __restrict__ src,
    const int* __restrict__ dst, float* __restrict__ agg, int E)
{
    int wave = (blockIdx.x * blockDim.x + threadIdx.x) >> 5;
    int lane = threadIdx.x & 31;
    if (wave >= E) return;
    int s = src[wave];
    int d = dst[wave];
    const float4* sp = (const float4*)(h + (size_t)s * GCN_H);
    float4 v = sp[lane];
    float* dp = agg + (size_t)d * GCN_H + lane * 4;
    atomicAdd(dp + 0, v.x);
    atomicAdd(dp + 1, v.y);
    atomicAdd(dp + 2, v.z);
    atomicAdd(dp + 3, v.w);
}

// ---------------------------------------------------------------------------
// Fragment builders.
// A (16x32 bf16): lane = row (lane&15), khalf = lane>>4.
//   element e -> k = base + (e<8 ? e : e+8), base = chunk*32 + khalf*8
// B (32x16 bf16) mirrors A with lane = column; k contiguous per column after
//   the LDS transpose, so the same run pattern applies.
// ---------------------------------------------------------------------------
__device__ inline v16bf gcn_afrag(const float* __restrict__ rowp, int base) {
    v16bf f;
#pragma unroll
    for (int e = 0; e < 8; ++e) f[e] = (__bf16)rowp[base + e];
#pragma unroll
    for (int e = 0; e < 8; ++e) f[8 + e] = (__bf16)rowp[base + 16 + e];
    return f;
}

__device__ inline v16bf gcn_bfrag(const __bf16* __restrict__ colp, int base) {
    v16bf f;
#pragma unroll
    for (int e = 0; e < 8; ++e) f[e] = colp[base + e];
#pragma unroll
    for (int e = 0; e < 8; ++e) f[8 + e] = colp[base + 16 + e];
    return f;
}

// ---------------------------------------------------------------------------
// Fused GCN layer GEMM: out = relu(A1@W + bW) + relu(A2@R + bR)
// A1,A2: [N,128] f32.  W,R: [128,128] f32 row-major.  out: [N,128] f32.
// Block = 256 threads = 8 waves; each wave owns one 16-row tile and sweeps
// all 8 column tiles; K=128 -> 4 chunks of 32 per WMMA accumulator.
// All 8 B-fragments of a column tile are preloaded into distinct registers
// before the WMMA burst so the ds loads clause up behind one s_wait_dscnt
// and the 8 WMMAs issue back-to-back (no WAR NOPs on the B operands).
// ---------------------------------------------------------------------------
__global__ void __launch_bounds__(256) gcn_gemm_fused(
    const float* __restrict__ A1, const float* __restrict__ A2,
    const float* __restrict__ W,  const float* __restrict__ bW,
    const float* __restrict__ R,  const float* __restrict__ bR,
    float* __restrict__ out, int nRowTiles)
{
    __shared__ __bf16 ldsW[GCN_H * LDS_K];
    __shared__ __bf16 ldsR[GCN_H * LDS_K];
    __shared__ float  ldsBW[GCN_H];
    __shared__ float  ldsBR[GCN_H];

    int t = threadIdx.x;
    // Stage weights into LDS, transposed (column-major, k contiguous) + bf16.
    for (int i = t; i < GCN_H * GCN_H; i += 256) {
        int k = i >> 7;
        int n = i & 127;
        ldsW[n * LDS_K + k] = (__bf16)W[i];
        ldsR[n * LDS_K + k] = (__bf16)R[i];
    }
    if (t < GCN_H) { ldsBW[t] = bW[t]; ldsBR[t] = bR[t]; }
    __syncthreads();

    int wave    = t >> 5;
    int lane    = t & 31;
    int rowTile = blockIdx.x * 8 + wave;
    if (rowTile >= nRowTiles) return;           // wave-uniform: EXEC stays full

    int m     = lane & 15;                      // row within tile / col within tile
    int khalf = lane >> 4;
    int row   = (rowTile << 4) + m;

    // Preload all A fragments (both operand matrices, 4 k-chunks each).
    const float* p1 = A1 + (size_t)row * GCN_H;
    const float* p2 = A2 + (size_t)row * GCN_H;
    v16bf a1[4], a2[4];
#pragma unroll
    for (int c = 0; c < 4; ++c) {
        int base = c * 32 + khalf * 8;
        a1[c] = gcn_afrag(p1, base);
        a2[c] = gcn_afrag(p2, base);
    }

    int rowBase = rowTile << 4;
#pragma unroll
    for (int ct = 0; ct < 8; ++ct) {
        const __bf16* wcol = ldsW + (ct * 16 + m) * LDS_K;
        const __bf16* rcol = ldsR + (ct * 16 + m) * LDS_K;

        // Preload the whole column tile's B fragments (8 x v16bf).
        v16bf bw[4], br[4];
#pragma unroll
        for (int c = 0; c < 4; ++c) {
            int base = c * 32 + khalf * 8;
            bw[c] = gcn_bfrag(wcol, base);
            br[c] = gcn_bfrag(rcol, base);
        }

        v8f acc1 = {};
        v8f acc2 = {};
#pragma unroll
        for (int c = 0; c < 4; ++c) {
            acc1 = __builtin_amdgcn_wmma_f32_16x16x32_bf16(
                false, a1[c], false, bw[c], (short)0, acc1, false, false);
            acc2 = __builtin_amdgcn_wmma_f32_16x16x32_bf16(
                false, a2[c], false, br[c], (short)0, acc2, false, false);
        }

        int col  = ct * 16 + m;
        float bwv = ldsBW[col];
        float brv = ldsBR[col];
#pragma unroll
        for (int g2 = 0; g2 < 8; ++g2) {
            int r2 = rowBase + g2 + khalf * 8;   // C/D layout: VGPR g -> M = g + 8*khalf
            float v = fmaxf(acc1[g2] + bwv, 0.0f) + fmaxf(acc2[g2] + brv, 0.0f);
            out[(size_t)r2 * GCN_H + col] = v;
        }
    }
}

// ---------------------------------------------------------------------------
// Readout: gate = sigmoid(h2 . w_gate + b); hsum += gate*h2 (atomic f32 add);
// hmax via unsigned atomicMax on bit pattern (h2 >= 0 so this is exact).
// One wave32 per node.
// ---------------------------------------------------------------------------
__global__ void __launch_bounds__(256) gcn_readout(
    const float* __restrict__ h2, const int* __restrict__ gids,
    const float* __restrict__ w_gate, const float* __restrict__ b_gate,
    float* __restrict__ hsum, unsigned* __restrict__ hmax, int N)
{
    int wave = (blockIdx.x * blockDim.x + threadIdx.x) >> 5;
    int lane = threadIdx.x & 31;
    if (wave >= N) return;

    const float4* hp = (const float4*)(h2 + (size_t)wave * GCN_H);
    const float4* wp = (const float4*)w_gate;
    float4 hv = hp[lane];
    float4 wv = wp[lane];
    float p = hv.x * wv.x + hv.y * wv.y + hv.z * wv.z + hv.w * wv.w;
#pragma unroll
    for (int off = 16; off > 0; off >>= 1) p += __shfl_xor(p, off, 32);
    float gate = 1.0f / (1.0f + expf(-(p + b_gate[0])));

    int g = gids[wave];
    float*    sp = hsum + (size_t)g * GCN_H + lane * 4;
    unsigned* mp = hmax + (size_t)g * GCN_H + lane * 4;
    atomicAdd(sp + 0, gate * hv.x);
    atomicAdd(sp + 1, gate * hv.y);
    atomicAdd(sp + 2, gate * hv.z);
    atomicAdd(sp + 3, gate * hv.w);
    atomicMax(mp + 0, __float_as_uint(hv.x));
    atomicMax(mp + 1, __float_as_uint(hv.y));
    atomicMax(mp + 2, __float_as_uint(hv.z));
    atomicMax(mp + 3, __float_as_uint(hv.w));
}

// ---------------------------------------------------------------------------
// Predictor MLP: z = relu([hsum,hmax] @ Wp1 + bp1); eval-BN; out = z @ Wp2 + bp2
// One block (128 threads) per graph; tiny GEMV, VALU is fine.
// ---------------------------------------------------------------------------
__global__ void __launch_bounds__(128) gcn_predict(
    const float* __restrict__ hsum, const float* __restrict__ hmax,
    const float* __restrict__ Wp1,  const float* __restrict__ bp1,
    const float* __restrict__ gamma, const float* __restrict__ beta,
    const float* __restrict__ rmean, const float* __restrict__ rvar,
    const float* __restrict__ Wp2,  const float* __restrict__ bp2,
    float* __restrict__ out)
{
    int g = blockIdx.x;
    int j = threadIdx.x;
    const float* hs = hsum + (size_t)g * GCN_H;
    const float* hm = hmax + (size_t)g * GCN_H;

    float z = bp1[j];
#pragma unroll 4
    for (int i = 0; i < GCN_H; ++i) z += hs[i] * Wp1[i * GCN_H + j];
#pragma unroll 4
    for (int i = 0; i < GCN_H; ++i) z += hm[i] * Wp1[(GCN_H + i) * GCN_H + j];
    z = fmaxf(z, 0.0f);
    z = (z - rmean[j]) * rsqrtf(rvar[j] + 1e-5f) * gamma[j] + beta[j];

    __shared__ float red[128];
    red[j] = z * Wp2[j];
    __syncthreads();
#pragma unroll
    for (int s = 64; s > 0; s >>= 1) {
        if (j < s) red[j] += red[j + s];
        __syncthreads();
    }
    if (j == 0) out[g] = red[0] + bp2[0];
}

// ---------------------------------------------------------------------------
// Launcher
// ---------------------------------------------------------------------------
extern "C" void kernel_launch(void* const* d_in, const int* in_sizes, int n_in,
                              void* d_out, int out_size, void* d_ws, size_t ws_size,
                              hipStream_t stream) {
    const float* x      = (const float*)d_in[0];
    const int*   src    = (const int*)  d_in[1];
    const int*   dst    = (const int*)  d_in[2];
    const int*   gids   = (const int*)  d_in[3];
    const float* W1     = (const float*)d_in[4];
    const float* b1     = (const float*)d_in[5];
    const float* R1     = (const float*)d_in[6];
    const float* rb1    = (const float*)d_in[7];
    const float* W2     = (const float*)d_in[8];
    const float* b2     = (const float*)d_in[9];
    const float* R2     = (const float*)d_in[10];
    const float* rb2    = (const float*)d_in[11];
    const float* w_gate = (const float*)d_in[12];
    const float* b_gate = (const float*)d_in[13];
    const float* Wp1    = (const float*)d_in[14];
    const float* bp1    = (const float*)d_in[15];
    const float* gamma  = (const float*)d_in[16];
    const float* beta   = (const float*)d_in[17];
    const float* rmean  = (const float*)d_in[18];
    const float* rvar   = (const float*)d_in[19];
    const float* Wp2    = (const float*)d_in[20];
    const float* bp2    = (const float*)d_in[21];
    float*       out    = (float*)d_out;

    const int N = in_sizes[0] / GCN_H;   // 40000
    const int E = in_sizes[1];           // 640000
    const int G = out_size;              // 256

    // Workspace carve-up (all offsets 256B-aligned).
    char*     ws   = (char*)d_ws;
    float*    agg  = (float*)(ws);                         // N*H f32 = 20.48 MB
    float*    h1   = (float*)(ws + 20480000);              // N*H f32
    float*    h2   = (float*)(ws + 40960000);              // N*H f32
    float*    hsum = (float*)(ws + 61440000);              // G*H f32
    unsigned* hmax = (unsigned*)(ws + 61440000 + 131072);  // G*H u32 (float bits)

    const int NH       = N * GCN_H;
    const int rowTiles = N / 16;                 // 2500
    dim3 blk256(256), blk128(128);
    dim3 zeroGrid((NH + 255) / 256);
    dim3 edgeGrid((E * 32 + 255) / 256);
    dim3 gemmGrid((rowTiles + 7) / 8);
    dim3 nodeGrid((N * 32 + 255) / 256);

    // ---- Layer 1 ----
    gcn_zero_f32<<<zeroGrid, blk256, 0, stream>>>(agg, NH);
    gcn_edge_scatter<<<edgeGrid, blk256, 0, stream>>>(x, src, dst, agg, E);
    gcn_gemm_fused<<<gemmGrid, blk256, 0, stream>>>(agg, x, W1, b1, R1, rb1, h1, rowTiles);

    // ---- Layer 2 ----
    gcn_zero_f32<<<zeroGrid, blk256, 0, stream>>>(agg, NH);
    gcn_edge_scatter<<<edgeGrid, blk256, 0, stream>>>(h1, src, dst, agg, E);
    gcn_gemm_fused<<<gemmGrid, blk256, 0, stream>>>(agg, h1, W2, b2, R2, rb2, h2, rowTiles);

    // ---- Readout ----
    gcn_zero_f32<<<dim3((2 * G * GCN_H + 255) / 256), blk256, 0, stream>>>(hsum, 2 * G * GCN_H);
    gcn_readout<<<nodeGrid, blk256, 0, stream>>>(h2, gids, w_gate, b_gate, hsum, hmax, N);

    // ---- Predictor ----
    gcn_predict<<<dim3(G), blk128, 0, stream>>>(hsum, (const float*)hmax, Wp1, bp1,
                                                gamma, beta, rmean, rvar, Wp2, bp2, out);
}
